// TGATModel_27041114095889
// MI455X (gfx1250) — matrix operand order repaired and checked by reading, not compiled, not run on hardware
//
#include <hip/hip_runtime.h>
#include <hip/hip_bf16.h>

typedef __attribute__((ext_vector_type(16))) __bf16 v16bf;
typedef __attribute__((ext_vector_type(8)))  __bf16 v8bf;
typedef __attribute__((ext_vector_type(8)))  float  v8f;

#define HEADS 4
#define FW 128            // feature width H*C for both layers
#define NEG_SLOPE 0.2f
#define EPSV 1e-16f
#define RPB 4             // 16-row blocks per workgroup in the GEMM (64 rows)

// ---- order-preserving float <-> uint map for atomic segment-max ----
__device__ __forceinline__ unsigned f2ord(float f) {
  unsigned u = __float_as_uint(f);
  return (u & 0x80000000u) ? ~u : (u | 0x80000000u);
}
__device__ __forceinline__ float ord2f(unsigned u) {
  return __uint_as_float((u & 0x80000000u) ? (u & 0x7fffffffu) : ~u);
}

// =====================================================================
// Pack X[n,K] f32 -> A-fragments bf16, WMMA-ready:
//   frag(rb,kb) lane l holds, contiguously, X[rb*16 + l%16][kb*32 + kbase + {0..7, 16..23}]
//   with kbase = (l<16 ? 0 : 8).  One thread per (rb,kb,lane): 32B store.
// Rows beyond n (incl. padded rowblocks) are zero-filled so the GEMM
// K-loop is branch-free (WMMA needs EXEC all-ones).
// =====================================================================
__global__ void pack_a_bf16(const float* __restrict__ X, __bf16* __restrict__ Af,
                            int n, int K, int rbPad) {
  const int nkb = K >> 5;
  const long t = blockIdx.x * (long)blockDim.x + threadIdx.x;
  if (t >= (long)rbPad * nkb * 32) return;
  const int lane = (int)(t & 31);
  const int kb   = (int)((t >> 5) % nkb);
  const int rb   = (int)((t >> 5) / nkb);
  const int m    = rb * 16 + (lane & 15);
  const int kbase = kb * 32 + ((lane < 16) ? 0 : 8);

  v8bf lo, hi;
  if (m < n) {
    const float* xp = X + (long)m * K + kbase;
#pragma unroll
    for (int j = 0; j < 8; ++j) { lo[j] = (__bf16)xp[j]; hi[j] = (__bf16)xp[16 + j]; }
  } else {
#pragma unroll
    for (int j = 0; j < 8; ++j) { lo[j] = (__bf16)0.f; hi[j] = (__bf16)0.f; }
  }
  v8bf* dst = (v8bf*)(Af + (t << 4));
  dst[0] = lo; dst[1] = hi;
}

// =====================================================================
// Pack W[K,128] f32 -> B-fragments bf16, WMMA-ready:
//   frag(kb,tile) lane l holds W[kb*32 + kbase + {0..7, 16..23}][tile*16 + l%16],
//   kbase = (l<16 ? 0 : 8).  One thread per (kb,tile,lane).
// =====================================================================
__global__ void pack_b_bf16(const float* __restrict__ W, __bf16* __restrict__ Bf,
                            int K) {
  const int nkb = K >> 5;
  const int t = blockIdx.x * blockDim.x + threadIdx.x;
  if (t >= nkb * 8 * 32) return;
  const int lane = t & 31;
  const int tile = (t >> 5) & 7;
  const int kb   = t >> 8;
  const int col  = tile * 16 + (lane & 15);
  const int krow = kb * 32 + ((lane < 16) ? 0 : 8);

  v8bf lo, hi;
#pragma unroll
  for (int j = 0; j < 8; ++j) {
    lo[j] = (__bf16)W[(long)(krow + j) * FW + col];
    hi[j] = (__bf16)W[(long)(krow + 16 + j) * FW + col];
  }
  v8bf* dst = (v8bf*)(Bf + ((long)t << 4));
  dst[0] = lo; dst[1] = hi;
}

// =====================================================================
// GEMM on packed fragments: Hout[n,128] = X @ W.
// Block = 128 threads (4 waves) -> 64x128 output tile.
// Wave w owns column tiles 2w, 2w+1.  Per K-step: 2 B loads (L2-resident),
// 4 A loads (streamed), 8 WMMAs.  Fully unrolled via template NKB.
// =====================================================================
template <int NKB>
__global__ __launch_bounds__(128)
void gemm_frag_wmma(const __bf16* __restrict__ Af, const __bf16* __restrict__ Bf,
                    float* __restrict__ Hout, int n) {
  const int wave = threadIdx.x >> 5;
  const int lane = threadIdx.x & 31;
  const int rb0  = blockIdx.x * RPB;

  const v16bf* __restrict__ Afv = (const v16bf*)Af;
  const v16bf* __restrict__ Bfv = (const v16bf*)Bf;

  v8f acc[RPB][2];
#pragma unroll
  for (int r = 0; r < RPB; ++r) { acc[r][0] = (v8f){}; acc[r][1] = (v8f){}; }

#pragma unroll
  for (int kb = 0; kb < NKB; ++kb) {
    const v16bf b0 = Bfv[(kb * 8 + wave * 2) * 32 + lane];
    const v16bf b1 = Bfv[(kb * 8 + wave * 2 + 1) * 32 + lane];
#pragma unroll
    for (int r = 0; r < RPB; ++r) {
      const v16bf a = Afv[(((long)(rb0 + r) * NKB + kb) * 32) + lane];
      acc[r][0] = __builtin_amdgcn_wmma_f32_16x16x32_bf16(false, a, false, b0,
                                                          (short)0, acc[r][0], false, false);
      acc[r][1] = __builtin_amdgcn_wmma_f32_16x16x32_bf16(false, a, false, b1,
                                                          (short)0, acc[r][1], false, false);
    }
  }

  // C/D layout: vgpr j, lanes<16 -> M=j ; lanes>=16 -> M=8+j ; N = lane%16
  const int nn = lane & 15;
  const int mb = (lane < 16) ? 0 : 8;
#pragma unroll
  for (int r = 0; r < RPB; ++r) {
#pragma unroll
    for (int j = 0; j < 8; ++j) {
      const int row = (rb0 + r) * 16 + mb + j;
      if (row < n) {
        Hout[(long)row * FW + wave * 32 + nn]      = acc[r][0][j];
        Hout[(long)row * FW + wave * 32 + 16 + nn] = acc[r][1][j];
      }
    }
  }
}

// =====================================================================
// Per-node attention scalars: Ssrc[n,g]=<h[n,g,:],a_src[g,:]>, same Sdst
// =====================================================================
__global__ void node_scalars(const float* __restrict__ Hm,
                             const float* __restrict__ a_src,
                             const float* __restrict__ a_dst,
                             float* __restrict__ Ssrc, float* __restrict__ Sdst,
                             long nh) {
  long t = blockIdx.x * (long)blockDim.x + threadIdx.x;
  if (t >= nh) return;
  long node = t >> 2;
  int  g    = (int)(t & 3);
  const float* hp = Hm + node * FW + g * 32;
  const float* as = a_src + g * 32;
  const float* ad = a_dst + g * 32;
  float ss = 0.f, sd = 0.f;
#pragma unroll
  for (int c = 0; c < 32; ++c) { float h = hp[c]; ss += h * as[c]; sd += h * ad[c]; }
  Ssrc[t] = ss; Sdst[t] = sd;
}

// weae[layer][d][g] = sum_c We[d, g*32+c] * ae[g, c]   (rank-collapse of e.ae)
__global__ void prep_weae(const float* __restrict__ We1, const float* __restrict__ ae1,
                          const float* __restrict__ We2, const float* __restrict__ ae2,
                          float* __restrict__ weae) {
  int t = threadIdx.x;            // 128 threads: 2 layers x 16 d x 4 g
  int layer = t >> 6;
  int i = t & 63;
  int d = i >> 2, g = i & 3;
  const float* We = layer ? We2 : We1;
  const float* ae = layer ? ae2 : ae1;
  float s = 0.f;
#pragma unroll
  for (int c = 0; c < 32; ++c) s += We[d * FW + g * 32 + c] * ae[g * 32 + c];
  weae[layer * 64 + d * 4 + g] = s;
}

__global__ void fill_u32(unsigned* __restrict__ p, unsigned v, long nelem) {
  long i = blockIdx.x * (long)blockDim.x + threadIdx.x;
  if (i < nelem) p[i] = v;
}

// =====================================================================
// Edge pass 1: leaky-relu logits -> ws ; atomic segment-max per (dst,head)
// =====================================================================
__global__ void edge_logit_max(const int* __restrict__ ei, long E,
                               const float* __restrict__ ea,
                               const float* __restrict__ weae,
                               const float* __restrict__ Ssrc,
                               const float* __restrict__ Sdst,
                               float* __restrict__ logit,
                               unsigned* __restrict__ Mmax) {
  long e = blockIdx.x * (long)blockDim.x + threadIdx.x;
  if (e >= E) return;
  const int s = ei[e], d = ei[E + e];
  float el[4] = {0.f, 0.f, 0.f, 0.f};
#pragma unroll
  for (int k = 0; k < 16; ++k) {
    float v = ea[e * 16 + k];
#pragma unroll
    for (int g = 0; g < 4; ++g) el[g] += v * weae[k * 4 + g];
  }
#pragma unroll
  for (int g = 0; g < 4; ++g) {
    float l = Ssrc[(long)s * 4 + g] + Sdst[(long)d * 4 + g] + el[g];
    l = (l > 0.f) ? l : NEG_SLOPE * l;
    logit[e * 4 + g] = l;
    atomicMax(&Mmax[(long)d * 4 + g], f2ord(l));
  }
}

// Edge pass 2: ex = exp(logit - m[dst]) -> ws (in place); atomic sum per (dst,head)
__global__ void edge_exp_sum(const int* __restrict__ ei, long E,
                             float* __restrict__ logit,
                             const unsigned* __restrict__ Mmax,
                             float* __restrict__ Ssum) {
  long e = blockIdx.x * (long)blockDim.x + threadIdx.x;
  if (e >= E) return;
  const int d = ei[E + e];
#pragma unroll
  for (int g = 0; g < 4; ++g) {
    float m  = ord2f(Mmax[(long)d * 4 + g]);
    float ex = __expf(logit[e * 4 + g] - m);
    logit[e * 4 + g] = ex;
    unsafeAtomicAdd(&Ssum[(long)d * 4 + g], ex);
  }
}

// Edge pass 3: one wave per edge; lane owns 4 channels (head = lane/8);
// Out[dst] += alpha * h[src]  via hardware f32 global atomics.
__global__ __launch_bounds__(256)
void edge_aggregate(const int* __restrict__ ei, long E,
                    const float* __restrict__ exv,
                    const float* __restrict__ Ssum,
                    const float* __restrict__ Hm,
                    float* __restrict__ Out) {
  long e = blockIdx.x * 8L + (threadIdx.x >> 5);
  if (e >= E) return;
  const int lane = threadIdx.x & 31;
  const int s = ei[e], d = ei[E + e];
  const int g = lane >> 3;
  float alpha = exv[e * 4 + g] / (Ssum[(long)d * 4 + g] + EPSV);
  const float4 h = *(const float4*)(Hm + (long)s * FW + lane * 4);
  float* op = Out + (long)d * FW + lane * 4;
  unsafeAtomicAdd(op + 0, alpha * h.x);
  unsafeAtomicAdd(op + 1, alpha * h.y);
  unsafeAtomicAdd(op + 2, alpha * h.z);
  unsafeAtomicAdd(op + 3, alpha * h.w);
}

// epilogue: A[i] += b[i%128]; optional relu (between layers)
__global__ void bias_act(float* __restrict__ A, const float* __restrict__ b,
                         long n128, int relu) {
  long i = blockIdx.x * (long)blockDim.x + threadIdx.x;
  if (i >= n128) return;
  float v = A[i] + b[i & 127];
  if (relu) v = fmaxf(v, 0.f);
  A[i] = v;
}

// =====================================================================
extern "C" void kernel_launch(void* const* d_in, const int* in_sizes, int n_in,
                              void* d_out, int out_size, void* d_ws, size_t ws_size,
                              hipStream_t stream) {
  const float* x   = (const float*)d_in[0];
  const int*   ei  = (const int*)d_in[1];     // edge_index [2,E] (int32)
  const float* ea  = (const float*)d_in[2];   // [E,16]
  const float* W1  = (const float*)d_in[3];
  const float* as1 = (const float*)d_in[4];
  const float* ad1 = (const float*)d_in[5];
  const float* We1 = (const float*)d_in[6];
  const float* ae1 = (const float*)d_in[7];
  const float* b1  = (const float*)d_in[8];
  const float* W2  = (const float*)d_in[9];
  const float* as2 = (const float*)d_in[10];
  const float* ad2 = (const float*)d_in[11];
  const float* We2 = (const float*)d_in[12];
  const float* ae2 = (const float*)d_in[13];
  const float* b2  = (const float*)d_in[14];

  const long N = in_sizes[0] / 256;
  const long E = in_sizes[1] / 2;

  const int rowblocks = (int)((N + 15) / 16);
  const int rbPad     = ((rowblocks + RPB - 1) / RPB) * RPB;

  // carve aligned scratch
  char* wsp = (char*)d_ws;
  auto carve = [&](size_t bytes) -> void* {
    void* p = wsp; wsp += (bytes + 255) & ~(size_t)255; return p;
  };
  float*    Hbuf = (float*)carve((size_t)N * FW * 4);    // h of current layer
  float*    Xbuf = (float*)carve((size_t)N * FW * 4);    // layer-1 agg -> relu -> layer-2 input
  float*    logw = (float*)carve((size_t)E * 4 * 4);     // logits, then exp values
  float*    Ssrc = (float*)carve((size_t)N * 4 * 4);
  float*    Sdst = (float*)carve((size_t)N * 4 * 4);
  unsigned* Mmax = (unsigned*)carve((size_t)N * 4 * 4);
  float*    Ssum = (float*)carve((size_t)N * 4 * 4);
  float*    weae = (float*)carve(2 * 64 * 4);
  __bf16*   Af   = (__bf16*)carve((size_t)rbPad * 16 * 256 * 2);  // packed A (max K=256)
  __bf16*   Bf1  = (__bf16*)carve((size_t)256 * FW * 2);          // packed W1
  __bf16*   Bf2  = (__bf16*)carve((size_t)128 * FW * 2);          // packed W2

  const long nh = N * 4;
  const unsigned ordNegInf = 0x007FFFFFu;  // f2ord(-inf)
  const unsigned gGemm = (unsigned)(rbPad / RPB);
  const unsigned gNH   = (unsigned)((nh + 255) / 256);
  const unsigned gE    = (unsigned)((E + 255) / 256);
  const unsigned gE8   = (unsigned)((E + 7) / 8);
  const unsigned gNF   = (unsigned)((N * FW + 255) / 256);
  const long paThreads1 = (long)rbPad * (256 / 32) * 32;
  const long paThreads2 = (long)rbPad * (128 / 32) * 32;

  prep_weae<<<1, 128, 0, stream>>>(We1, ae1, We2, ae2, weae);
  pack_b_bf16<<<(256 * 8 * 32 / 32 + 255) / 256 + 1, 256, 0, stream>>>(W1, Bf1, 256);
  pack_b_bf16<<<(128 * 8 * 32 / 32 + 255) / 256 + 1, 256, 0, stream>>>(W2, Bf2, 128);

  // ---------------- layer 1 ----------------
  hipMemsetAsync(Xbuf, 0, (size_t)N * FW * 4, stream);
  hipMemsetAsync(Ssum, 0, (size_t)N * 4 * 4, stream);
  fill_u32<<<gNH, 256, 0, stream>>>(Mmax, ordNegInf, nh);

  pack_a_bf16<<<(unsigned)((paThreads1 + 255) / 256), 256, 0, stream>>>(x, Af, (int)N, 256, rbPad);
  gemm_frag_wmma<8><<<gGemm, 128, 0, stream>>>(Af, Bf1, Hbuf, (int)N);
  node_scalars<<<gNH, 256, 0, stream>>>(Hbuf, as1, ad1, Ssrc, Sdst, nh);
  edge_logit_max<<<gE, 256, 0, stream>>>(ei, E, ea, weae, Ssrc, Sdst, logw, Mmax);
  edge_exp_sum<<<gE, 256, 0, stream>>>(ei, E, logw, Mmax, Ssum);
  edge_aggregate<<<gE8, 256, 0, stream>>>(ei, E, logw, Ssum, Hbuf, Xbuf);
  bias_act<<<gNF, 256, 0, stream>>>(Xbuf, b1, N * FW, 1);   // + b1, relu

  // ---------------- layer 2 ----------------
  float* Out = (float*)d_out;
  hipMemsetAsync(Out, 0, (size_t)N * FW * 4, stream);
  hipMemsetAsync(Ssum, 0, (size_t)N * 4 * 4, stream);
  fill_u32<<<gNH, 256, 0, stream>>>(Mmax, ordNegInf, nh);

  pack_a_bf16<<<(unsigned)((paThreads2 + 255) / 256), 256, 0, stream>>>(Xbuf, Af, (int)N, 128, rbPad);
  gemm_frag_wmma<4><<<gGemm, 128, 0, stream>>>(Af, Bf2, Hbuf, (int)N);
  node_scalars<<<gNH, 256, 0, stream>>>(Hbuf, as2, ad2, Ssrc, Sdst, nh);
  edge_logit_max<<<gE, 256, 0, stream>>>(ei, E, ea, weae + 64, Ssrc, Sdst, logw, Mmax);
  edge_exp_sum<<<gE, 256, 0, stream>>>(ei, E, logw, Mmax, Ssum);
  edge_aggregate<<<gE8, 256, 0, stream>>>(ei, E, logw, Ssum, Hbuf, Out);
  bias_act<<<gNF, 256, 0, stream>>>(Out, b2, N * FW, 0);    // + b2
}
